// mGCNConv_33028298506591
// MI455X (gfx1250) — matrix operand order repaired and checked by reading, not compiled, not run on hardware
//
#include <hip/hip_runtime.h>

// mGCNConv for MI455X (gfx1250): bf16 WMMA GEMMs + L2-resident edge scatter.
// Pipeline: cvt/transpose weights -> dim_repr GEMM (leaky, bf16 out) ->
// attention (small GEMMs + reduce + softmax) -> concat init (across+bias) ->
// per-dim {hw GEMM, degree, rsqrt, self-loop, edge scatter-atomics} ->
// concat->bf16 -> final GEMM (bias+leaky) to f32 d_out.

typedef __bf16 bf16;
typedef __attribute__((ext_vector_type(16))) __bf16 v16bf;
typedef __attribute__((ext_vector_type(8)))  __bf16 v8bf;
typedef __attribute__((ext_vector_type(8)))  float  v8f;

union V16BF { v16bf v; v8bf h[2]; };

#define IN_DIM  512
#define H_DIM   256
#define OUT_DIM 512
#define NDIMS   4
#define NEG_SLOPE 0.01f

__device__ __forceinline__ float leaky_f(float v) { return v > 0.f ? v : NEG_SLOPE * v; }

// ---------------------------------------------------------------------------
// Tiled bf16 WMMA GEMM: C[M,N] = act(A[M,K] @ B[K,N] + bias)
// A: row-major bf16.  Bt: B transposed, row-major [N,K] bf16 (so each output
// column's K values are contiguous -> two b128 loads per fragment per lane).
// Block = 128 threads = 4 waves; block tile 128x64; wave tile 32x64:
// 2 A fragments x 4 B fragments -> 8 v_wmma_f32_16x16x32_bf16 per K-step,
// B fragments reused across both row strips (8 WMMA per 6 fragment loads).
// ---------------------------------------------------------------------------
template<bool LEAKY, bool BIAS, bool OUT_F32, bool OUT_BF16>
__global__ __launch_bounds__(128) void wmma_gemm_bf16(
    const bf16* __restrict__ A, const bf16* __restrict__ Bt,
    const float* __restrict__ bias,
    float* __restrict__ Cf, bf16* __restrict__ Cb,
    int M, int N, int K)
{
    const int lane = threadIdx.x & 31;
    const int wave = threadIdx.x >> 5;   // 0..3
    const int l16  = lane & 15;
    const int hi   = lane >> 4;          // 0 or 1
    const int tileN = blockIdx.x * 64;
    const int row0  = blockIdx.y * 128 + wave * 32;

    // A fragment (16x32 bf16): lane row = row0 + strip*16 + l16;
    // lanes 0-15 hold K=[0,16), lanes 16-31 hold K=[16,32) (contiguous).
    int ar0 = row0 + l16;      if (ar0 >= M) ar0 = M - 1;   // clamp OOB reads
    int ar1 = row0 + 16 + l16; if (ar1 >= M) ar1 = M - 1;
    const bf16* Ap0 = A + (size_t)ar0 * K + hi * 16;
    const bf16* Ap1 = A + (size_t)ar1 * K + hi * 16;

    // B fragment (32x16 bf16): lane col = tileN + t*16 + l16;
    // lanes 0-15 hold K=[0,16), lanes 16-31 hold K=[16,32) (contiguous in Bt).
    const bf16* Bp0 = Bt + (size_t)(tileN +  0 + l16) * K + hi * 16;
    const bf16* Bp1 = Bt + (size_t)(tileN + 16 + l16) * K + hi * 16;
    const bf16* Bp2 = Bt + (size_t)(tileN + 32 + l16) * K + hi * 16;
    const bf16* Bp3 = Bt + (size_t)(tileN + 48 + l16) * K + hi * 16;

    v8f acc[8] = {};   // [strip0: t0..t3, strip1: t0..t3]
    for (int k0 = 0; k0 < K; k0 += 32) {
        V16BF a0, a1;
        a0.h[0] = *(const v8bf*)(Ap0 + k0);
        a0.h[1] = *(const v8bf*)(Ap0 + k0 + 8);
        a1.h[0] = *(const v8bf*)(Ap1 + k0);
        a1.h[1] = *(const v8bf*)(Ap1 + k0 + 8);
        __builtin_prefetch(Ap0 + k0 + 32, 0, 1);   // global_prefetch_b8
        __builtin_prefetch(Ap1 + k0 + 32, 0, 1);
        V16BF b0, b1, b2, b3;
        b0.h[0] = *(const v8bf*)(Bp0 + k0); b0.h[1] = *(const v8bf*)(Bp0 + k0 + 8);
        b1.h[0] = *(const v8bf*)(Bp1 + k0); b1.h[1] = *(const v8bf*)(Bp1 + k0 + 8);
        b2.h[0] = *(const v8bf*)(Bp2 + k0); b2.h[1] = *(const v8bf*)(Bp2 + k0 + 8);
        b3.h[0] = *(const v8bf*)(Bp3 + k0); b3.h[1] = *(const v8bf*)(Bp3 + k0 + 8);
        acc[0] = __builtin_amdgcn_wmma_f32_16x16x32_bf16(false, a0.v, false, b0.v,
                                                         (short)0, acc[0], false, false);
        acc[1] = __builtin_amdgcn_wmma_f32_16x16x32_bf16(false, a0.v, false, b1.v,
                                                         (short)0, acc[1], false, false);
        acc[2] = __builtin_amdgcn_wmma_f32_16x16x32_bf16(false, a0.v, false, b2.v,
                                                         (short)0, acc[2], false, false);
        acc[3] = __builtin_amdgcn_wmma_f32_16x16x32_bf16(false, a0.v, false, b3.v,
                                                         (short)0, acc[3], false, false);
        acc[4] = __builtin_amdgcn_wmma_f32_16x16x32_bf16(false, a1.v, false, b0.v,
                                                         (short)0, acc[4], false, false);
        acc[5] = __builtin_amdgcn_wmma_f32_16x16x32_bf16(false, a1.v, false, b1.v,
                                                         (short)0, acc[5], false, false);
        acc[6] = __builtin_amdgcn_wmma_f32_16x16x32_bf16(false, a1.v, false, b2.v,
                                                         (short)0, acc[6], false, false);
        acc[7] = __builtin_amdgcn_wmma_f32_16x16x32_bf16(false, a1.v, false, b3.v,
                                                         (short)0, acc[7], false, false);
    }

    // C/D layout: VGPR v holds row = v (lanes 0-15) / 8+v (lanes 16-31), col = l16.
    #pragma unroll
    for (int s = 0; s < 2; ++s) {
        const int rbase = row0 + s * 16 + hi * 8;
        #pragma unroll
        for (int t = 0; t < 4; ++t) {
            const int c = tileN + t * 16 + l16;
            const float bv = BIAS ? bias[c] : 0.f;
            const v8f av = acc[s * 4 + t];
            #pragma unroll
            for (int v = 0; v < 8; ++v) {
                const int r = rbase + v;
                if (r < M) {
                    float val = av[v] + bv;
                    if (LEAKY) val = leaky_f(val);
                    if (OUT_F32)  Cf[(size_t)r * N + c] = val;
                    if (OUT_BF16) Cb[(size_t)r * N + c] = (bf16)val;
                }
            }
        }
    }
}

// ---------------------------------------------------------------------------
// Elementwise / prep kernels
// ---------------------------------------------------------------------------
__global__ void cvt_f32_bf16(const float* __restrict__ in, bf16* __restrict__ out, size_t n) {
    size_t i = (size_t)blockIdx.x * blockDim.x + threadIdx.x;
    if (i < n) out[i] = (bf16)in[i];
}

__global__ void transpose_f32_bf16(const float* __restrict__ in, bf16* __restrict__ out,
                                   int R, int C) {   // out[c*R + r] = in[r*C + c]
    size_t i = (size_t)blockIdx.x * blockDim.x + threadIdx.x;
    if (i >= (size_t)R * C) return;
    int r = (int)(i / C), c = (int)(i % C);
    out[(size_t)c * R + r] = (bf16)in[i];
}

__global__ void deg_init(float* __restrict__ deg, int N) {
    int i = blockIdx.x * blockDim.x + threadIdx.x;
    if (i < N) deg[i] = 1.0f;   // self-loop contribution
}

__global__ void deg_count(const int* __restrict__ dst, float* __restrict__ deg, int E) {
    int e = blockIdx.x * blockDim.x + threadIdx.x;
    if (e < E) atomicAdd(&deg[dst[e]], 1.0f);
}

__global__ void deg_rsqrt(float* __restrict__ deg, int N) {
    int i = blockIdx.x * blockDim.x + threadIdx.x;
    if (i < N) { float d = deg[i]; deg[i] = d > 0.f ? rsqrtf(d) : 0.f; }
}

// attention[a,b] = sum_{j,i,k} proj[a,j,i] * bil[i,k] * proj[b,j,k]
// tmpA[a] = proj[a] @ bil done via WMMA GEMM; here the final dot.
__global__ void att_reduce(const float* __restrict__ tmpA, const float* __restrict__ proj,
                           float* __restrict__ att) {
    const int a = blockIdx.x >> 2, b = blockIdx.x & 3;
    const int n = IN_DIM * H_DIM;
    const float* pa = tmpA + (size_t)a * n;
    const float* pb = proj + (size_t)b * n;
    float s = 0.f;
    for (int i = threadIdx.x; i < n; i += blockDim.x) s += pa[i] * pb[i];
    __shared__ float red[256];
    red[threadIdx.x] = s; __syncthreads();
    for (int st = 128; st > 0; st >>= 1) {
        if (threadIdx.x < st) red[threadIdx.x] += red[threadIdx.x + st];
        __syncthreads();
    }
    if (threadIdx.x == 0) att[a * 4 + b] = red[0];
}

__global__ void att_softmax(float* __restrict__ att) {   // softmax over axis 0 (a)
    int b = threadIdx.x;
    if (b < 4) {
        float m = -1e30f;
        for (int a = 0; a < 4; ++a) m = fmaxf(m, att[a * 4 + b]);
        float e[4], s = 0.f;
        for (int a = 0; a < 4; ++a) { e[a] = __expf(att[a * 4 + b] - m); s += e[a]; }
        for (int a = 0; a < 4; ++a) att[a * 4 + b] = e[a] / s;
    }
}

// concat[n, b*H + h] = 0.5 * sum_a attsm[a,b] * dim_repr[a,n,h] + 0.5 * gcn_b[b,h]
__global__ void concat_init(const bf16* __restrict__ drep, const float* __restrict__ attsm,
                            const float* __restrict__ gcnb, float* __restrict__ concat, int N) {
    size_t i = (size_t)blockIdx.x * blockDim.x + threadIdx.x;
    if (i >= (size_t)N * H_DIM) return;
    int n = (int)(i / H_DIM), h = (int)(i % H_DIM);
    float v[NDIMS];
    #pragma unroll
    for (int a = 0; a < NDIMS; ++a)
        v[a] = (float)drep[((size_t)a * N + n) * H_DIM + h];
    #pragma unroll
    for (int b = 0; b < NDIMS; ++b) {
        float s = 0.f;
        #pragma unroll
        for (int a = 0; a < NDIMS; ++a) s += attsm[a * 4 + b] * v[a];
        concat[(size_t)n * (NDIMS * H_DIM) + b * H_DIM + h] = 0.5f * s + 0.5f * gcnb[b * H_DIM + h];
    }
}

// self-loop: concat[i, dOff+h] += 0.5 * dinv[i]^2 * hw[i,h]  (unique i -> plain add)
__global__ void self_scatter(const float* __restrict__ dinv, const float* __restrict__ hw,
                             float* __restrict__ concat, int dOff) {
    int i = blockIdx.x, h = threadIdx.x;
    float w = 0.5f * dinv[i] * dinv[i];
    concat[(size_t)i * (NDIMS * H_DIM) + dOff + h] += w * hw[(size_t)i * H_DIM + h];
}

// edges: concat[dst, dOff+h] += 0.5 * dinv[src]*dinv[dst] * hw[src,h]
// hw (51 MB) stays resident in the 192 MB L2 -> gather + atomic both L2-speed.
__global__ void edge_scatter(const int* __restrict__ src, const int* __restrict__ dst,
                             const float* __restrict__ dinv, const float* __restrict__ hw,
                             float* __restrict__ concat, int dOff) {
    int e = blockIdx.x, h = threadIdx.x;
    int s = src[e], t = dst[e];
    float w = 0.5f * dinv[s] * dinv[t];
    atomicAdd(concat + (size_t)t * (NDIMS * H_DIM) + dOff + h, w * hw[(size_t)s * H_DIM + h]);
}

// ---------------------------------------------------------------------------
extern "C" void kernel_launch(void* const* d_in, const int* in_sizes, int n_in,
                              void* d_out, int out_size, void* d_ws, size_t ws_size,
                              hipStream_t stream) {
    const float* x    = (const float*)d_in[0];
    const int*   eidx = (const int*)d_in[1];     // [4, 2, E]
    const float* proj = (const float*)d_in[2];   // [4, 512, 256]
    const float* gcnW = (const float*)d_in[3];   // [4, 256, 256]
    const float* gcnb = (const float*)d_in[4];   // [4, 256]
    const float* bil  = (const float*)d_in[5];   // [256, 256]
    const float* invW = (const float*)d_in[6];   // [1024, 512]
    const float* invb = (const float*)d_in[7];   // [512]
    float* out = (float*)d_out;

    const int N = in_sizes[0] / IN_DIM;
    const int E = in_sizes[1] / (NDIMS * 2);

    // workspace carve-out (256B aligned)
    char* ws = (char*)d_ws;
    size_t off = 0;
    auto alloc = [&](size_t bytes) -> void* {
        void* p = ws + off;
        off = (off + bytes + 255) & ~(size_t)255;
        return p;
    };
    bf16* xb     = (bf16*)alloc((size_t)N * IN_DIM * 2);
    bf16* projT  = (bf16*)alloc((size_t)NDIMS * H_DIM * IN_DIM * 2);   // B^T per dim
    bf16* projA  = (bf16*)alloc((size_t)NDIMS * IN_DIM * H_DIM * 2);   // A for attention
    bf16* bilT   = (bf16*)alloc((size_t)H_DIM * H_DIM * 2);
    bf16* gcnWT  = (bf16*)alloc((size_t)NDIMS * H_DIM * H_DIM * 2);
    bf16* invWT  = (bf16*)alloc((size_t)OUT_DIM * NDIMS * H_DIM * 2);
    bf16* drep   = (bf16*)alloc((size_t)NDIMS * N * H_DIM * 2);        // leaky(x@proj) bf16
    float* hw    = (float*)alloc((size_t)N * H_DIM * 4);               // reused per dim
    float* dinv  = (float*)alloc((size_t)N * 4);                       // reused per dim
    float* tmpA  = (float*)alloc((size_t)NDIMS * IN_DIM * H_DIM * 4);
    float* att   = (float*)alloc(64);
    float* concat = (float*)alloc((size_t)N * NDIMS * H_DIM * 4);
    bf16* concatB = (bf16*)alloc((size_t)N * NDIMS * H_DIM * 2);
    (void)ws_size; (void)n_in; (void)out_size;

    const int mtiles = (N + 127) / 128;

    // --- weight / input conversion -----------------------------------------
    {
        size_t n0 = (size_t)N * IN_DIM;
        cvt_f32_bf16<<<(unsigned)((n0 + 255) / 256), 256, 0, stream>>>(x, xb, n0);
        size_t n1 = (size_t)NDIMS * IN_DIM * H_DIM;
        cvt_f32_bf16<<<(unsigned)((n1 + 255) / 256), 256, 0, stream>>>(proj, projA, n1);
        for (int d = 0; d < NDIMS; ++d)
            transpose_f32_bf16<<<(IN_DIM * H_DIM + 255) / 256, 256, 0, stream>>>(
                proj + (size_t)d * IN_DIM * H_DIM, projT + (size_t)d * H_DIM * IN_DIM, IN_DIM, H_DIM);
        transpose_f32_bf16<<<(H_DIM * H_DIM + 255) / 256, 256, 0, stream>>>(bil, bilT, H_DIM, H_DIM);
        for (int d = 0; d < NDIMS; ++d)
            transpose_f32_bf16<<<(H_DIM * H_DIM + 255) / 256, 256, 0, stream>>>(
                gcnW + (size_t)d * H_DIM * H_DIM, gcnWT + (size_t)d * H_DIM * H_DIM, H_DIM, H_DIM);
        transpose_f32_bf16<<<(NDIMS * H_DIM * OUT_DIM + 255) / 256, 256, 0, stream>>>(
            invW, invWT, NDIMS * H_DIM, OUT_DIM);
    }

    // --- dim_repr[d] = leaky(x @ proj[d]) -> bf16 --------------------------
    for (int d = 0; d < NDIMS; ++d)
        wmma_gemm_bf16<true, false, false, true><<<dim3(H_DIM / 64, mtiles), 128, 0, stream>>>(
            xb, projT + (size_t)d * H_DIM * IN_DIM, nullptr,
            nullptr, drep + (size_t)d * N * H_DIM, N, H_DIM, IN_DIM);

    // --- attention = softmax_a( <proj[a]@bil, proj[b]> ) -------------------
    for (int a = 0; a < NDIMS; ++a)
        wmma_gemm_bf16<false, false, true, false><<<dim3(H_DIM / 64, IN_DIM / 128), 128, 0, stream>>>(
            projA + (size_t)a * IN_DIM * H_DIM, bilT, nullptr,
            tmpA + (size_t)a * IN_DIM * H_DIM, nullptr, IN_DIM, H_DIM, H_DIM);
    att_reduce<<<16, 256, 0, stream>>>(tmpA, proj, att);
    att_softmax<<<1, 32, 0, stream>>>(att);

    // --- concat <- 0.5*across + 0.5*bias -----------------------------------
    concat_init<<<(unsigned)(((size_t)N * H_DIM + 255) / 256), 256, 0, stream>>>(
        drep, att, gcnb, concat, N);

    // --- per-dim GCN: hw GEMM, norms, scatter-add (+= 0.5*within) ----------
    for (int d = 0; d < NDIMS; ++d) {
        const int* src = eidx + (size_t)d * 2 * E;
        const int* dst = src + E;
        wmma_gemm_bf16<false, false, true, false><<<dim3(H_DIM / 64, mtiles), 128, 0, stream>>>(
            drep + (size_t)d * N * H_DIM, gcnWT + (size_t)d * H_DIM * H_DIM, nullptr,
            hw, nullptr, N, H_DIM, H_DIM);
        deg_init<<<(N + 255) / 256, 256, 0, stream>>>(dinv, N);
        deg_count<<<(E + 255) / 256, 256, 0, stream>>>(dst, dinv, E);
        deg_rsqrt<<<(N + 255) / 256, 256, 0, stream>>>(dinv, N);
        self_scatter<<<N, H_DIM, 0, stream>>>(dinv, hw, concat, d * H_DIM);
        edge_scatter<<<E, H_DIM, 0, stream>>>(src, dst, dinv, hw, concat, d * H_DIM);
    }

    // --- final: out = leaky(concat @ inv_W + inv_b) ------------------------
    {
        size_t nc = (size_t)N * NDIMS * H_DIM;
        cvt_f32_bf16<<<(unsigned)((nc + 255) / 256), 256, 0, stream>>>(concat, concatB, nc);
        wmma_gemm_bf16<true, true, true, false><<<dim3(OUT_DIM / 64, mtiles), 128, 0, stream>>>(
            concatB, invWT, invb, out, nullptr, N, OUT_DIM, NDIMS * H_DIM);
    }
}
